// AttentionHead_80917183856668
// MI455X (gfx1250) — matrix operand order repaired and checked by reading, not compiled, not run on hardware
//
#include <hip/hip_runtime.h>
#include <stdint.h>

// ---------------- problem dimensions ----------------
#define Bb 8
#define Ss 4096
#define DM 1024
#define DKk 128
#define DVv 128
#define MF 256
#define BS (Bb*Ss)                      // 32768 rows
#define EPSF 1e-6f
#define QK_SCALE 0.2973017787506803f    // 128^-0.25
#define RSQRT_M 0.0625f                 // 256^-0.5

typedef unsigned short u16;
typedef __attribute__((ext_vector_type(16))) __bf16        v16bf;
typedef __attribute__((ext_vector_type(8)))  float         v8f;
typedef __attribute__((ext_vector_type(4)))  unsigned int  u32x4;
typedef __attribute__((ext_vector_type(2)))  unsigned int  u32x2;
typedef __attribute__((ext_vector_type(4)))  float         f32x4;
typedef __attribute__((ext_vector_type(4)))  u16           u16x4;

union FragU { v16bf v; u32x4 q[2]; };

// ---------------- CDNA5 async global->LDS staging ----------------
// Signature discovered from clang diagnostics:
//   (gcc-vector-int AS(1)*, gcc-vector-int AS(3)*, int offset, int cpol)
typedef int gv2i __attribute__((vector_size(8)));
typedef int gv4i __attribute__((vector_size(16)));
#define AS1 __attribute__((address_space(1)))
#define AS3 __attribute__((address_space(3)))

#if __has_builtin(__builtin_amdgcn_global_load_async_to_lds_b128) && \
    __has_builtin(__builtin_amdgcn_global_load_async_to_lds_b64)
#define HAVE_ASYNC 1
#else
#define HAVE_ASYNC 0
#endif

__device__ inline void async_b64(void* lds, const void* g) {
#if HAVE_ASYNC
  __builtin_amdgcn_global_load_async_to_lds_b64((AS1 gv2i*)g, (AS3 gv2i*)lds, 0, 0);
#else
  *(u32x2*)lds = *(const u32x2*)g;
#endif
}
__device__ inline void async_b128(void* lds, const void* g) {
#if HAVE_ASYNC
  __builtin_amdgcn_global_load_async_to_lds_b128((AS1 gv4i*)g, (AS3 gv4i*)lds, 0, 0);
#else
  *(u32x4*)lds = *(const u32x4*)g;
#endif
}
__device__ inline void async_wait_all() {
#if HAVE_ASYNC
#if __has_builtin(__builtin_amdgcn_s_wait_asynccnt)
  __builtin_amdgcn_s_wait_asynccnt(0);
#else
  asm volatile("s_wait_asynccnt 0x0" ::: "memory");
#endif
#endif
}

// ---------------- scalar helpers ----------------
__device__ inline float bf2f(u16 u) { return __uint_as_float(((unsigned int)u) << 16); }
__device__ inline u16 f2bf(float f) {                    // RNE truncation
  unsigned int u = __float_as_uint(f);
  u += 0x7FFFu + ((u >> 16) & 1u);
  return (u16)(u >> 16);
}
// monotone float<->uint mapping so unsigned atomicMax == float max (deterministic)
__device__ inline unsigned int encF(float f) {
  unsigned int u = __float_as_uint(f);
  return (u & 0x80000000u) ? ~u : (u | 0x80000000u);
}
__device__ inline float decF(unsigned int e) {
  unsigned int u = (e & 0x80000000u) ? (e & 0x7FFFFFFFu) : ~e;
  return __uint_as_float(u);
}

// ---------------- WMMA fragment loaders (CDNA5 wave32 layouts) ----------------
// A (16x32 bf16): row = lane&15; lanes 0-15 hold K runs [0..7],[16..23];
// lanes 16-31 hold [8..15],[24..31]  (ISA 7.12.2)
__device__ inline v16bf load_frag_a(const u16* base, int row0, int ld, int kb) {
  int lane = threadIdx.x & 31;
  const u16* p = base + (size_t)(row0 + (lane & 15)) * ld + kb + ((lane >> 4) << 3);
  FragU u;
  u.q[0] = *(const u32x4*)(p);
  u.q[1] = *(const u32x4*)(p + 16);
  return u.v;
}
// B (32x16 bf16) staged N-major in LDS: col = lane&15; lanes 0-15 hold K runs [0..7],[16..23];
// lanes 16-31 hold [8..15],[24..31]
__device__ inline v16bf load_frag_b(const u16* base, int n0, int ld, int kb) {
  int lane = threadIdx.x & 31;
  const u16* p = base + (size_t)(n0 + (lane & 15)) * ld + kb + ((lane >> 4) << 3);
  FragU u;
  u.q[0] = *(const u32x4*)(p);
  u.q[1] = *(const u32x4*)(p + 16);
  return u.v;
}
__device__ inline v8f wmma_bf16(v16bf a, v16bf b, v8f c) {
  return __builtin_amdgcn_wmma_f32_16x16x32_bf16(false, a, false, b, (short)0, c,
                                                 false, false);
}

// ---------------- K0: weight transpose + bf16 cast + init ----------------
__global__ __launch_bounds__(256) void prep_kernel(
    const float* __restrict__ Wq, const float* __restrict__ Wk,
    const float* __restrict__ Wv, const float* __restrict__ om,
    u16* __restrict__ wtQ, u16* __restrict__ wtK, u16* __restrict__ wtV,
    u16* __restrict__ omBf, unsigned int* __restrict__ gmaxEnc) {
  const int WSZ = DM * DKk;            // 131072
  const int TOT = 3 * WSZ + MF * DKk + 1;
  for (int idx = blockIdx.x * blockDim.x + threadIdx.x; idx < TOT;
       idx += gridDim.x * blockDim.x) {
    if (idx < 3 * WSZ) {
      int sel = idx / WSZ, e = idx % WSZ;
      int n = e >> 10, k = e & 1023;                     // wt[n][k] = W[k][n]
      const float* W = (sel == 0) ? Wq : (sel == 1) ? Wk : Wv;
      u16* wt = (sel == 0) ? wtQ : (sel == 1) ? wtK : wtV;
      wt[e] = f2bf(W[(size_t)k * DKk + n]);
    } else if (idx < 3 * WSZ + MF * DKk) {
      int e = idx - 3 * WSZ;
      omBf[e] = f2bf(om[e]);
    } else {
      *gmaxEnc = 0u;                                     // maps below any real value
    }
  }
}

// ---------------- K1: fused q/k/v projection (x read ONCE, 12 WMMA/K-step) ----
__global__ __launch_bounds__(256) void proj_kernel(
    const float* __restrict__ x,
    const u16* __restrict__ wtQ, const u16* __restrict__ wtK,
    const u16* __restrict__ wtV,
    const float* __restrict__ bq, const float* __restrict__ bk,
    const float* __restrict__ bv,
    u16* __restrict__ qbf, u16* __restrict__ kbf, u16* __restrict__ vT) {
  __shared__ __align__(16) u16 sX[2][64 * 32];           // 8 KB
  __shared__ __align__(16) u16 sW[2][3][128 * 32];       // 48 KB
  const int t = threadIdx.x, wave = t >> 5, lane = t & 31;
  const int rowBase = blockIdx.x * 64;

  v8f acc[3][4];
  for (int s = 0; s < 3; ++s)
    for (int r = 0; r < 4; ++r)
      for (int i = 0; i < 8; ++i) acc[s][r][i] = 0.f;

  auto stage = [&](int buf, int kb) {
#pragma unroll
    for (int i = 0; i < 2; ++i) {                        // x tile f32 -> bf16 (VALU path)
      int e = t + i * 256, r = e >> 3, c4 = e & 7;
      const float* gp = &x[(size_t)(rowBase + r) * DM + kb + c4 * 4];
      f32x4 xv = *(const f32x4*)gp;
      u16x4 o;
      o[0] = f2bf(xv[0]); o[1] = f2bf(xv[1]); o[2] = f2bf(xv[2]); o[3] = f2bf(xv[3]);
      *(u16x4*)&sX[buf][r * 32 + c4 * 4] = o;
      if (kb + 32 < DM) __builtin_prefetch(gp + 32, 0, 1);   // global_prefetch_b8
    }
#pragma unroll
    for (int i = 0; i < 2; ++i) {                        // 3 weight tiles, async b128
      int e = t + i * 256, n = e >> 2, c8 = e & 3;
      int lo = n * 32 + c8 * 8;
      size_t go = (size_t)n * DM + kb + c8 * 8;
      async_b128(&sW[buf][0][lo], &wtQ[go]);
      async_b128(&sW[buf][1][lo], &wtK[go]);
      async_b128(&sW[buf][2][lo], &wtV[go]);
    }
  };

  stage(0, 0);
  for (int kb = 0; kb < DM; kb += 32) {
    int cur = (kb >> 5) & 1;
    async_wait_all();
    __syncthreads();
    if (kb + 32 < DM) stage(cur ^ 1, kb + 32);
    v16bf bQ = load_frag_b(sW[cur][0], wave * 16, 32, 0);
    v16bf bK = load_frag_b(sW[cur][1], wave * 16, 32, 0);
    v16bf bV = load_frag_b(sW[cur][2], wave * 16, 32, 0);
#pragma unroll
    for (int r = 0; r < 4; ++r) {
      v16bf a = load_frag_a(sX[cur], r * 16, 32, 0);
      acc[0][r] = wmma_bf16(a, bQ, acc[0][r]);
      acc[1][r] = wmma_bf16(a, bK, acc[1][r]);
      acc[2][r] = wmma_bf16(a, bV, acc[2][r]);
    }
  }

  const int col = wave * 16 + (lane & 15);
  const float bQc = bq[col], bKc = bk[col], bVc = bv[col];
#pragma unroll
  for (int r = 0; r < 4; ++r)
#pragma unroll
    for (int i = 0; i < 8; ++i) {
      int tr = r * 16 + i + ((lane >> 4) << 3);
      int rowG = rowBase + tr;
      qbf[(size_t)rowG * DKk + col] = f2bf((acc[0][r][i] + bQc) * QK_SCALE);
      kbf[(size_t)rowG * DKk + col] = f2bf((acc[1][r][i] + bKc) * QK_SCALE);
      int b = rowG >> 12, s = rowG & (Ss - 1);
      vT[((size_t)(b * DVv + col)) * Ss + s] = f2bf(acc[2][r][i] + bVc);  // [B][DV][S]
    }
}

// ---------------- K2: pq/pk = qk @ omega^T - 0.5*||.||^2 (WMMA, K=128) -------
__global__ __launch_bounds__(256) void feat_kernel(
    const u16* __restrict__ qbf, const u16* __restrict__ kbf,
    const u16* __restrict__ omBf,
    float* __restrict__ pq, float* __restrict__ pk,
    unsigned int* __restrict__ gmaxEnc) {
  __shared__ __align__(16) u16 sA[64 * 128];             // 16 KB
  __shared__ __align__(16) u16 sB[128 * 128];            // 32 KB
  __shared__ float sPart[256];
  __shared__ float sNorm[64];
  const int t = threadIdx.x, wave = t >> 5, lane = t & 31;
  const int rowBase = blockIdx.x * 64;
  const int nHalf = blockIdx.y;                          // feature-column half
  const int sel = blockIdx.z;                            // 0:q 1:k
  const u16* src = sel ? kbf : qbf;
  float* dst = sel ? pk : pq;

#pragma unroll
  for (int i = 0; i < 4; ++i) {                          // q/k tile 64x128, async
    int e = t + i * 256, r = e >> 4, c8 = e & 15;
    async_b128(&sA[r * 128 + c8 * 8], &src[(size_t)(rowBase + r) * DKk + c8 * 8]);
  }
#pragma unroll
  for (int i = 0; i < 8; ++i) {                          // omega chunk 128x128, async
    int e = t + i * 256, n = e >> 4, c8 = e & 15;
    async_b128(&sB[n * 128 + c8 * 8],
               &omBf[(size_t)(nHalf * 128 + n) * DKk + c8 * 8]);
  }
  async_wait_all();
  __syncthreads();
  {                                                      // per-row squared norms
    int row = t >> 2, part = t & 3;
    float s = 0.f;
    for (int j = 0; j < 32; ++j) {
      float q = bf2f(sA[row * 128 + part * 32 + j]);
      s += q * q;
    }
    sPart[t] = s;
  }
  __syncthreads();
  if (t < 64)
    sNorm[t] = sPart[4 * t] + sPart[4 * t + 1] + sPart[4 * t + 2] + sPart[4 * t + 3];
  __syncthreads();

  v8f acc[4];
  for (int r = 0; r < 4; ++r)
    for (int i = 0; i < 8; ++i) acc[r][i] = 0.f;
#pragma unroll
  for (int kb = 0; kb < DKk; kb += 32) {
    v16bf bfr = load_frag_b(sB, wave * 16, 128, kb);
#pragma unroll
    for (int r = 0; r < 4; ++r) {
      v16bf afr = load_frag_a(sA, r * 16, 128, kb);
      acc[r] = wmma_bf16(afr, bfr, acc[r]);
    }
  }

  float tmax = -3.4e38f;
  const int m = nHalf * 128 + wave * 16 + (lane & 15);
#pragma unroll
  for (int r = 0; r < 4; ++r) {
#pragma unroll
    for (int i = 0; i < 8; ++i) {
      int tr = r * 16 + i + ((lane >> 4) << 3);
      float p = acc[r][i] - 0.5f * sNorm[tr];
      dst[(size_t)(rowBase + tr) * MF + m] = p;
      tmax = fmaxf(tmax, p);
    }
  }
  if (sel) {                                             // global key stabilizer
    for (int o = 16; o > 0; o >>= 1) tmax = fmaxf(tmax, __shfl_xor(tmax, o));
    if (lane == 0) atomicMax(gmaxEnc, encF(tmax));
  }
}

// ---------------- K3: exp feature maps ----------------
__global__ __launch_bounds__(256) void phi_kernel(
    const float* __restrict__ pq, const float* __restrict__ pk,
    const unsigned int* __restrict__ gmaxEnc,
    u16* __restrict__ phiQ, u16* __restrict__ phiKT) {
  const int wave = threadIdx.x >> 5, lane = threadIdx.x & 31;
  const int row = blockIdx.x * 8 + wave;
  if (blockIdx.y == 0) {                                 // phi_q: per-row max
    float p[8], mx = -3.4e38f;
#pragma unroll
    for (int j = 0; j < 8; ++j) {
      p[j] = pq[(size_t)row * MF + lane + 32 * j];
      mx = fmaxf(mx, p[j]);
    }
    for (int o = 16; o > 0; o >>= 1) mx = fmaxf(mx, __shfl_xor(mx, o));
#pragma unroll
    for (int j = 0; j < 8; ++j)
      phiQ[(size_t)row * MF + lane + 32 * j] = f2bf(expf(p[j] - mx) * RSQRT_M);
  } else {                                               // phi_k: global max, transposed
    float g = decF(*gmaxEnc);
    int b = row >> 12, s = row & (Ss - 1);
#pragma unroll
    for (int j = 0; j < 8; ++j) {
      int m = lane + 32 * j;
      float p = pk[(size_t)row * MF + m];
      phiKT[((size_t)(b * MF + m)) * Ss + s] = f2bf(expf(p - g) * RSQRT_M);
    }
  }
}

// ---------------- K5: ksum[b][m] = sum_s phi_k ----------------
__global__ __launch_bounds__(256) void ksum_kernel(const u16* __restrict__ phiKT,
                                                   float* __restrict__ ksum) {
  const int wave = threadIdx.x >> 5, lane = threadIdx.x & 31;
  const int row = blockIdx.x * 8 + wave;                 // b*MF + m
  float s = 0.f;
  for (int j = 0; j < Ss / 32; ++j) s += bf2f(phiKT[(size_t)row * Ss + lane + 32 * j]);
  for (int o = 16; o > 0; o >>= 1) s += __shfl_xor(s, o);
  if (lane == 0) ksum[row] = s;
}

// ---------------- K4: kv = phi_k^T @ v (WMMA, K=4096, double-buffered) -------
__global__ __launch_bounds__(256) void kv_kernel(const u16* __restrict__ phiKT,
                                                 const u16* __restrict__ vT,
                                                 u16* __restrict__ kvT) {
  __shared__ __align__(16) u16 sA[2][64 * 32];           // 8 KB
  __shared__ __align__(16) u16 sB[2][128 * 32];          // 16 KB
  const int t = threadIdx.x, wave = t >> 5, lane = t & 31;
  const int mBase = blockIdx.x * 64;
  const int b = blockIdx.y;
  v8f acc[4];
  for (int r = 0; r < 4; ++r)
    for (int i = 0; i < 8; ++i) acc[r][i] = 0.f;

  auto stage = [&](int buf, int sb) {
    {
      int r = t >> 2, c8 = t & 3;                        // 256 b128 units
      async_b128(&sA[buf][r * 32 + c8 * 8],
                 &phiKT[((size_t)(b * MF + mBase + r)) * Ss + sb + c8 * 8]);
    }
#pragma unroll
    for (int i = 0; i < 2; ++i) {                        // 512 b128 units
      int e = t + i * 256, n = e >> 2, c8 = e & 3;
      async_b128(&sB[buf][n * 32 + c8 * 8],
                 &vT[((size_t)(b * DVv + n)) * Ss + sb + c8 * 8]);
    }
  };

  stage(0, 0);
  for (int sb = 0; sb < Ss; sb += 32) {
    int cur = (sb >> 5) & 1;
    async_wait_all();
    __syncthreads();
    if (sb + 32 < Ss) stage(cur ^ 1, sb + 32);
    v16bf bfr = load_frag_b(sB[cur], wave * 16, 32, 0);
#pragma unroll
    for (int r = 0; r < 4; ++r) {
      v16bf afr = load_frag_a(sA[cur], r * 16, 32, 0);
      acc[r] = wmma_bf16(afr, bfr, acc[r]);
    }
  }
  const int d = wave * 16 + (lane & 15);
#pragma unroll
  for (int r = 0; r < 4; ++r)
#pragma unroll
    for (int i = 0; i < 8; ++i) {
      int mr = mBase + r * 16 + i + ((lane >> 4) << 3);
      kvT[(size_t)(b * DVv + d) * MF + mr] = f2bf(acc[r][i]);   // transposed for B-frags
    }
}

// ---------------- K6: num = phi_q @ kv, den, divide (double-buffered) --------
__global__ __launch_bounds__(256) void out_kernel(const u16* __restrict__ phiQ,
                                                  const u16* __restrict__ kvT,
                                                  const float* __restrict__ ksum,
                                                  float* __restrict__ out) {
  __shared__ __align__(16) u16 sA[2][64 * 32];
  __shared__ __align__(16) u16 sB[2][128 * 32];
  __shared__ float sPart[256];
  __shared__ float sDen[64];
  const int t = threadIdx.x, wave = t >> 5, lane = t & 31;
  const int rowBase = blockIdx.x * 64;
  const int b = rowBase >> 12;
  v8f acc[4];
  for (int r = 0; r < 4; ++r)
    for (int i = 0; i < 8; ++i) acc[r][i] = 0.f;

  auto stage = [&](int buf, int mb) {
    {
      int r = t >> 2, c8 = t & 3;
      async_b128(&sA[buf][r * 32 + c8 * 8],
                 &phiQ[(size_t)(rowBase + r) * MF + mb + c8 * 8]);
    }
#pragma unroll
    for (int i = 0; i < 2; ++i) {
      int e = t + i * 256, n = e >> 2, c8 = e & 3;
      async_b128(&sB[buf][n * 32 + c8 * 8],
                 &kvT[(size_t)(b * DVv + n) * MF + mb + c8 * 8]);
    }
  };

  stage(0, 0);
  for (int mb = 0; mb < MF; mb += 32) {
    int cur = (mb >> 5) & 1;
    async_wait_all();
    __syncthreads();
    if (mb + 32 < MF) stage(cur ^ 1, mb + 32);
    v16bf bfr = load_frag_b(sB[cur], wave * 16, 32, 0);
#pragma unroll
    for (int r = 0; r < 4; ++r) {
      v16bf afr = load_frag_a(sA[cur], r * 16, 32, 0);
      acc[r] = wmma_bf16(afr, bfr, acc[r]);
    }
  }
  {                                                      // denominator per row
    int row = t >> 2, part = t & 3;
    float s = 0.f;
    for (int j = 0; j < 64; ++j) {
      int m = part * 64 + j;
      s += bf2f(phiQ[(size_t)(rowBase + row) * MF + m]) * ksum[b * MF + m];
    }
    sPart[t] = s;
  }
  __syncthreads();
  if (t < 64)
    sDen[t] = sPart[4 * t] + sPart[4 * t + 1] + sPart[4 * t + 2] + sPart[4 * t + 3] + EPSF;
  __syncthreads();
  const int d = wave * 16 + (lane & 15);
#pragma unroll
  for (int r = 0; r < 4; ++r)
#pragma unroll
    for (int i = 0; i < 8; ++i) {
      int tr = r * 16 + i + ((lane >> 4) << 3);
      out[(size_t)(rowBase + tr) * DVv + d] = acc[r][i] / sDen[tr];
    }
}

// ---------------- host-side orchestration ----------------
extern "C" void kernel_launch(void* const* d_in, const int* in_sizes, int n_in,
                              void* d_out, int out_size, void* d_ws, size_t ws_size,
                              hipStream_t stream) {
  (void)in_sizes; (void)n_in; (void)out_size; (void)ws_size;
  const float* x  = (const float*)d_in[0];
  const float* Wq = (const float*)d_in[1];
  const float* bq = (const float*)d_in[2];
  const float* Wk = (const float*)d_in[3];
  const float* bk = (const float*)d_in[4];
  const float* Wv = (const float*)d_in[5];
  const float* bv = (const float*)d_in[6];
  const float* om = (const float*)d_in[7];
  float* out = (float*)d_out;

  uint8_t* ws = (uint8_t*)d_ws;
  size_t off = 0;
  auto take = [&](size_t bytes) -> void* {
    void* p = ws + off;
    off += (bytes + 255) & ~(size_t)255;
    return p;
  };
  u16* wtQ   = (u16*)take((size_t)DM * DKk * 2);
  u16* wtK   = (u16*)take((size_t)DM * DKk * 2);
  u16* wtV   = (u16*)take((size_t)DM * DKk * 2);
  u16* omBf  = (u16*)take((size_t)MF * DKk * 2);
  u16* qbf   = (u16*)take((size_t)BS * DKk * 2);
  u16* kbf   = (u16*)take((size_t)BS * DKk * 2);
  u16* vT    = (u16*)take((size_t)Bb * DVv * Ss * 2);
  float* pqB = (float*)take((size_t)BS * MF * 4);
  float* pkB = (float*)take((size_t)BS * MF * 4);
  u16* phiQ  = (u16*)take((size_t)BS * MF * 2);
  u16* phiKT = (u16*)take((size_t)Bb * MF * Ss * 2);
  u16* kvT   = (u16*)take((size_t)Bb * DVv * MF * 2);
  float* ksum = (float*)take((size_t)Bb * MF * 4);
  unsigned int* gmax = (unsigned int*)take(256);

  prep_kernel<<<dim3(256), 256, 0, stream>>>(Wq, Wk, Wv, om, wtQ, wtK, wtV, omBf, gmax);
  proj_kernel<<<dim3(BS / 64), 256, 0, stream>>>(x, wtQ, wtK, wtV, bq, bk, bv,
                                                 qbf, kbf, vT);
  feat_kernel<<<dim3(BS / 64, 2, 2), 256, 0, stream>>>(qbf, kbf, omBf, pqB, pkB, gmax);
  phi_kernel<<<dim3(BS / 8, 2), 256, 0, stream>>>(pqB, pkB, gmax, phiQ, phiKT);
  ksum_kernel<<<dim3(Bb * MF / 8), 256, 0, stream>>>(phiKT, ksum);
  kv_kernel<<<dim3(MF / 64, Bb), 256, 0, stream>>>(phiKT, vT, kvT);
  out_kernel<<<dim3(BS / 64), 256, 0, stream>>>(phiQ, kvT, ksum, out);
}
